// AttentionSimple_18631568130548
// MI455X (gfx1250) — matrix operand order repaired and checked by reading, not compiled, not run on hardware
//
#include <hip/hip_runtime.h>
#include <hip/hip_bf16.h>
#include <cstdint>

// ---------------------------------------------------------------------------
// Problem constants (B=16, N=4096, C=768, H=12, D=64)
// ---------------------------------------------------------------------------
#define BATCH   16
#define SEQ     4096
#define CH      768
#define NH      12
#define HD      64
#define BN_ROWS (BATCH * SEQ)          // 65536
#define QKV_C   (3 * CH)               // 2304
#define SCALE_F 0.125f                 // 64^-0.5

typedef __attribute__((ext_vector_type(16))) __bf16 v16bf;
typedef __attribute__((ext_vector_type(8)))  __bf16 v8bf;
typedef __attribute__((ext_vector_type(8)))  float  v8f;
typedef __attribute__((ext_vector_type(4)))  int    v4i;

typedef v4i __attribute__((address_space(1))) gv4i;   // global int4
typedef v4i __attribute__((address_space(3))) lv4i;   // LDS int4

__device__ __forceinline__ float gelu_f(float x) {
    return 0.5f * x * (1.0f + erff(x * 0.70710678118654752f));
}

// ---------------------------------------------------------------------------
// CDNA5 async global->LDS copy (16B per lane), ASYNCcnt-tracked
// ---------------------------------------------------------------------------
__device__ __forceinline__ void async_cp16(const __bf16* g, __bf16* l) {
#if __has_builtin(__builtin_amdgcn_global_load_async_to_lds_b128)
    __builtin_amdgcn_global_load_async_to_lds_b128(
        (gv4i*)g, (lv4i*)l, 0, 0);
#else
    asm volatile("global_load_async_to_lds_b128 %0, %1, off"
                 :: "v"((uint32_t)(uintptr_t)l), "v"(g) : "memory");
#endif
}

__device__ __forceinline__ void wait_async0() {
#if __has_builtin(__builtin_amdgcn_s_wait_asynccnt)
    __builtin_amdgcn_s_wait_asynccnt(0);
#else
    asm volatile("s_wait_asynccnt 0" ::: "memory");
#endif
}

// ---------------------------------------------------------------------------
// fp32 -> bf16 elementwise convert (8 elems / thread, b128 store)
// ---------------------------------------------------------------------------
__global__ void __launch_bounds__(256) k_cvt_bf16(const float* __restrict__ in,
                                                  __bf16* __restrict__ out, int n) {
    int i = (blockIdx.x * 256 + threadIdx.x) * 8;
    if (i >= n) return;
    float4 f0 = *(const float4*)(in + i);
    float4 f1 = *(const float4*)(in + i + 4);
    v8bf o;
    o[0] = (__bf16)f0.x; o[1] = (__bf16)f0.y; o[2] = (__bf16)f0.z; o[3] = (__bf16)f0.w;
    o[4] = (__bf16)f1.x; o[5] = (__bf16)f1.y; o[6] = (__bf16)f1.z; o[7] = (__bf16)f1.w;
    *(v8bf*)(out + i) = o;
}

// fp32 [R x Cc] -> bf16 transposed [Cc x R] (weights only; tiny)
__global__ void __launch_bounds__(256) k_cvt_bf16_t(const float* __restrict__ in,
                                                    __bf16* __restrict__ out,
                                                    int R, int Cc) {
    int idx = blockIdx.x * 256 + threadIdx.x;
    if (idx >= R * Cc) return;
    int r = idx / Cc, c = idx % Cc;
    out[(size_t)c * R + r] = (__bf16)in[idx];
}

// ---------------------------------------------------------------------------
// Hot-path bf16 WMMA GEMM with async double-buffered LDS staging.
//   C[BM x BN block] = A[M,K] @ Bt[N,K]^T      (Bt is pre-transposed, row n
//   holds K contiguous) -> both tiles are straight 16B async copies.
//   8 waves, 4x2 layout, 64x64 per wave = 4x4 WMMA tiles.
// ---------------------------------------------------------------------------
template<int BM, int BN, bool OUT_F32>
__device__ __forceinline__ void gemm_async_dev(
    const __bf16* __restrict__ A, int lda,
    const __bf16* __restrict__ Bt, int ldbt,
    int K,
    __bf16* __restrict__ Cb, float* __restrict__ Cf, int ldc,
    const float* __restrict__ bias,
    int m0, int n0)
{
    constexpr int BK      = 32;
    constexpr int LDS_S   = BK + 8;           // 40 halfs = 80B stride, conflict-free
    constexpr int WAVES_M = 4, WAVES_N = 2;
    constexpr int WTM = BM / WAVES_M;         // 64
    constexpr int WTN = BN / WAVES_N;         // 64
    constexpr int TI  = WTM / 16;             // 4
    constexpr int TJ  = WTN / 16;             // 4

    __shared__ __align__(16) __bf16 As[2][BM * LDS_S];
    __shared__ __align__(16) __bf16 Bs[2][BN * LDS_S];

    const int tid  = threadIdx.x;
    const int lane = tid & 31;
    const int w    = tid >> 5;
    const int wm   = w % WAVES_M;
    const int wn   = w / WAVES_M;
    const int half = lane >> 4;
    const int l16  = lane & 15;

    v8f acc[TI][TJ] = {};

    auto stage = [&](int buf, int k0) {
        #pragma unroll
        for (int p = 0; p < (BM * 4) / 256; ++p) {
            int c = tid + p * 256;
            int r = c >> 2, ck = (c & 3) * 8;
            async_cp16(A + (size_t)(m0 + r) * lda + k0 + ck,
                       &As[buf][r * LDS_S + ck]);
        }
        #pragma unroll
        for (int p = 0; p < (BN * 4) / 256; ++p) {
            int c = tid + p * 256;
            int r = c >> 2, ck = (c & 3) * 8;
            async_cp16(Bt + (size_t)(n0 + r) * ldbt + k0 + ck,
                       &Bs[buf][r * LDS_S + ck]);
        }
    };

    stage(0, 0);
    wait_async0();
    __syncthreads();

    int buf = 0;
    for (int k0 = 0; k0 < K; k0 += BK) {
        const bool more = (k0 + BK) < K;
        if (more) stage(buf ^ 1, k0 + BK);   // overlap copy of tile k+1 with compute

        // B fragments once per K-step
        v16bf bfv[TJ];
        #pragma unroll
        for (int j = 0; j < TJ; j++) {
            int n = wn * WTN + j * 16 + l16;
            v8bf b0 = *(const v8bf*)&Bs[buf][n * LDS_S + half * 8];
            v8bf b1 = *(const v8bf*)&Bs[buf][n * LDS_S + 16 + half * 8];
            #pragma unroll
            for (int e = 0; e < 8; e++) { bfv[j][e] = b0[e]; bfv[j][8 + e] = b1[e]; }
        }
        // A fragment loaded just-in-time per i-row to bound register pressure
        #pragma unroll
        for (int i = 0; i < TI; i++) {
            int m = wm * WTM + i * 16 + l16;
            v8bf a0 = *(const v8bf*)&As[buf][m * LDS_S + half * 8];
            v8bf a1 = *(const v8bf*)&As[buf][m * LDS_S + 16 + half * 8];
            v16bf af;
            #pragma unroll
            for (int e = 0; e < 8; e++) { af[e] = a0[e]; af[8 + e] = a1[e]; }
            #pragma unroll
            for (int j = 0; j < TJ; j++)
                acc[i][j] = __builtin_amdgcn_wmma_f32_16x16x32_bf16(
                    false, af, false, bfv[j], (short)0, acc[i][j], false, false);
        }

        if (more) { wait_async0(); __syncthreads(); }
        buf ^= 1;
    }

    #pragma unroll
    for (int i = 0; i < TI; i++)
        #pragma unroll
        for (int j = 0; j < TJ; j++)
            #pragma unroll
            for (int r = 0; r < 8; r++) {
                int m = m0 + wm * WTM + i * 16 + half * 8 + r;
                int n = n0 + wn * WTN + j * 16 + l16;
                float v = acc[i][j][r];
                if constexpr (OUT_F32) Cf[(size_t)m * ldc + n] = v + bias[n];
                else                   Cb[(size_t)m * ldc + n] = (__bf16)v;
            }
}

// qkv = x @ w_qkv  (Wt pre-transposed [2304 x 768]) -> bf16 [65536 x 2304]
__global__ void __launch_bounds__(256, 1)
k_gemm_qkv(const __bf16* __restrict__ X, const __bf16* __restrict__ Wt,
           __bf16* __restrict__ Out) {
    gemm_async_dev<256, 128, false>(X, CH, Wt, CH, CH, Out, nullptr, QKV_C, nullptr,
                                    blockIdx.y * 256, blockIdx.x * 128);
}

// out = attn @ w_proj + b_proj  (Wt pre-transposed [768 x 768]) -> fp32
__global__ void __launch_bounds__(256, 1)
k_gemm_proj(const __bf16* __restrict__ Attn, const __bf16* __restrict__ Wt,
            float* __restrict__ Out, const float* __restrict__ bias) {
    gemm_async_dev<256, 128, true>(Attn, CH, Wt, CH, CH, nullptr, Out, CH, bias,
                                   blockIdx.y * 256, blockIdx.x * 128);
}

// ---------------------------------------------------------------------------
// Generic (register-staged) WMMA GEMM, used only for gelu(q) @ gkv (2% FLOPs):
// gelu applied to A during staging; B staged transposed in LDS.
// ---------------------------------------------------------------------------
template<int BM, int BN, int WAVES_M, int WAVES_N>
__device__ __forceinline__ void gemm_gelu_dev(
    const __bf16* __restrict__ A, int lda,
    const __bf16* __restrict__ B, int ldb,
    int K,
    __bf16* __restrict__ Cb, int ldc,
    int m0, int n0)
{
    constexpr int BK    = 32;
    constexpr int LDS_S = BK + 8;
    constexpr int WTM   = BM / WAVES_M;
    constexpr int WTN   = BN / WAVES_N;
    constexpr int TI    = WTM / 16;
    constexpr int TJ    = WTN / 16;

    __shared__ __align__(16) __bf16 As[BM * LDS_S];
    __shared__ __align__(16) __bf16 Bs[BN * LDS_S];

    const int tid  = threadIdx.x;
    const int lane = tid & 31;
    const int w    = tid >> 5;
    const int wm   = w % WAVES_M;
    const int wn   = w / WAVES_M;
    const int half = lane >> 4;
    const int l16  = lane & 15;

    v8f acc[TI][TJ] = {};

    for (int k0 = 0; k0 < K; k0 += BK) {
        #pragma unroll
        for (int p = 0; p < (BM * 4) / 256; ++p) {
            int c = tid + p * 256;
            int row = c >> 2, ck = (c & 3) * 8;
            v8bf val = *(const v8bf*)(A + (size_t)(m0 + row) * lda + k0 + ck);
            #pragma unroll
            for (int e = 0; e < 8; e++) val[e] = (__bf16)gelu_f((float)val[e]);
            *(v8bf*)&As[row * LDS_S + ck] = val;
        }
        constexpr int NCH = BN / 8;
        #pragma unroll
        for (int p = 0; p < (BK * NCH) / 256; ++p) {
            int c = tid + p * 256;
            int kr = c / NCH, nc = (c % NCH) * 8;
            v8bf val = *(const v8bf*)(B + (size_t)(k0 + kr) * ldb + n0 + nc);
            #pragma unroll
            for (int e = 0; e < 8; e++) Bs[(nc + e) * LDS_S + kr] = val[e];
        }
        __syncthreads();

        v16bf bfv[TJ];
        #pragma unroll
        for (int j = 0; j < TJ; j++) {
            int n = wn * WTN + j * 16 + l16;
            v8bf b0 = *(const v8bf*)&Bs[n * LDS_S + half * 8];
            v8bf b1 = *(const v8bf*)&Bs[n * LDS_S + 16 + half * 8];
            #pragma unroll
            for (int e = 0; e < 8; e++) { bfv[j][e] = b0[e]; bfv[j][8 + e] = b1[e]; }
        }
        #pragma unroll
        for (int i = 0; i < TI; i++) {
            int m = wm * WTM + i * 16 + l16;
            v8bf a0 = *(const v8bf*)&As[m * LDS_S + half * 8];
            v8bf a1 = *(const v8bf*)&As[m * LDS_S + 16 + half * 8];
            v16bf af;
            #pragma unroll
            for (int e = 0; e < 8; e++) { af[e] = a0[e]; af[8 + e] = a1[e]; }
            #pragma unroll
            for (int j = 0; j < TJ; j++)
                acc[i][j] = __builtin_amdgcn_wmma_f32_16x16x32_bf16(
                    false, af, false, bfv[j], (short)0, acc[i][j], false, false);
        }
        __syncthreads();
    }

    #pragma unroll
    for (int i = 0; i < TI; i++)
        #pragma unroll
        for (int j = 0; j < TJ; j++)
            #pragma unroll
            for (int r = 0; r < 8; r++) {
                int m = m0 + wm * WTM + i * 16 + half * 8 + r;
                int n = n0 + wn * WTN + j * 16 + l16;
                Cb[(size_t)m * ldc + n] = (__bf16)acc[i][j][r];
            }
}

// attn = gelu(q) @ gkv  per (b,h): [4096x64] @ [64x64] -> bf16 into [65536x768]
__global__ void __launch_bounds__(256, 1)
k_gemm_attn(const __bf16* __restrict__ QKV, const __bf16* __restrict__ GKV,
            __bf16* __restrict__ Attn) {
    int z = blockIdx.z; int b = z / NH, h = z % NH;
    const __bf16* A  = QKV + (size_t)b * SEQ * QKV_C + h * HD;       // q slice
    const __bf16* Bp = GKV + (size_t)z * HD * HD;
    __bf16*       C  = Attn + (size_t)b * SEQ * CH + h * HD;
    gemm_gelu_dev<128, 64, 8, 1>(A, QKV_C, Bp, HD, HD, C, CH, blockIdx.y * 128, 0);
}

// ---------------------------------------------------------------------------
// kv[b,h] = gelu( scale * k^T v ) : [64x64] per (b,h), K-reduction over 4096.
// ---------------------------------------------------------------------------
__global__ void __launch_bounds__(256, 1)
k_kv(const __bf16* __restrict__ QKV, __bf16* __restrict__ GKV) {
    constexpr int LDS_S = 40;
    __shared__ __align__(16) __bf16 KsT[HD * LDS_S];   // [d][n] transposed
    __shared__ __align__(16) __bf16 VsT[HD * LDS_S];   // [e][n] transposed

    const int z = blockIdx.x; const int b = z / NH, h = z % NH;
    const __bf16* kbase = QKV + (size_t)b * SEQ * QKV_C + CH     + h * HD;
    const __bf16* vbase = QKV + (size_t)b * SEQ * QKV_C + 2 * CH + h * HD;

    const int tid = threadIdx.x, lane = tid & 31, w = tid >> 5;
    const int half = lane >> 4, l16 = lane & 15;
    const int i0 = w >> 1;            // d-tile 0..3
    const int j0 = (w & 1) * 2;       // e-tile base

    v8f acc[2] = {};

    const int sn = tid >> 3;          // 0..31 (row n within tile)
    const int dc = (tid & 7) * 8;     // 0..56 (d-chunk)

    for (int n0 = 0; n0 < SEQ; n0 += 32) {
        v8bf kv_ = *(const v8bf*)(kbase + (size_t)(n0 + sn) * QKV_C + dc);
        v8bf vv_ = *(const v8bf*)(vbase + (size_t)(n0 + sn) * QKV_C + dc);
        #pragma unroll
        for (int e = 0; e < 8; e++) {
            KsT[(dc + e) * LDS_S + sn] = kv_[e];
            VsT[(dc + e) * LDS_S + sn] = vv_[e];
        }
        __syncthreads();

        v16bf af;
        {
            int d = i0 * 16 + l16;
            v8bf a0 = *(const v8bf*)&KsT[d * LDS_S + half * 8];
            v8bf a1 = *(const v8bf*)&KsT[d * LDS_S + 16 + half * 8];
            #pragma unroll
            for (int e = 0; e < 8; e++) { af[e] = a0[e]; af[8 + e] = a1[e]; }
        }
        #pragma unroll
        for (int jj = 0; jj < 2; jj++) {
            int ec = (j0 + jj) * 16 + l16;
            v16bf bfv;
            v8bf b0 = *(const v8bf*)&VsT[ec * LDS_S + half * 8];
            v8bf b1 = *(const v8bf*)&VsT[ec * LDS_S + 16 + half * 8];
            #pragma unroll
            for (int e = 0; e < 8; e++) { bfv[e] = b0[e]; bfv[8 + e] = b1[e]; }
            acc[jj] = __builtin_amdgcn_wmma_f32_16x16x32_bf16(
                false, af, false, bfv, (short)0, acc[jj], false, false);
        }
        __syncthreads();
    }

    #pragma unroll
    for (int jj = 0; jj < 2; jj++)
        #pragma unroll
        for (int r = 0; r < 8; r++) {
            int d = i0 * 16 + half * 8 + r;
            int e = (j0 + jj) * 16 + l16;
            float v = gelu_f(acc[jj][r] * SCALE_F);
            GKV[(size_t)z * HD * HD + d * HD + e] = (__bf16)v;
        }
}

// ---------------------------------------------------------------------------
// Host-side launcher
// ---------------------------------------------------------------------------
extern "C" void kernel_launch(void* const* d_in, const int* in_sizes, int n_in,
                              void* d_out, int out_size, void* d_ws, size_t ws_size,
                              hipStream_t stream) {
    const float* x      = (const float*)d_in[0];   // [16,4096,768]
    const float* w_qkv  = (const float*)d_in[1];   // [768,2304]
    const float* w_proj = (const float*)d_in[2];   // [768,768]
    const float* b_proj = (const float*)d_in[3];   // [768]
    float*       out    = (float*)d_out;           // [16,4096,768]

    char*  ws  = (char*)d_ws;
    size_t off = 0;
    auto carve = [&](size_t bytes) -> void* {
        void* p = ws + off;
        off += (bytes + 255) & ~(size_t)255;
        return p;
    };
    __bf16* xb    = (__bf16*)carve((size_t)BN_ROWS * CH * 2);
    __bf16* wqbT  = (__bf16*)carve((size_t)CH * QKV_C * 2);   // [2304 x 768]
    __bf16* wpbT  = (__bf16*)carve((size_t)CH * CH * 2);      // [768 x 768]
    __bf16* qkvb  = (__bf16*)carve((size_t)BN_ROWS * QKV_C * 2);
    __bf16* gkvb  = (__bf16*)carve((size_t)BATCH * NH * HD * HD * 2);
    __bf16* attnb = (__bf16*)carve((size_t)BN_ROWS * CH * 2);
    (void)ws_size; (void)n_in; (void)in_sizes; (void)out_size;

    // 1) fp32 -> bf16 converts (weights transposed so GEMM staging is copy-only)
    {
        int nx = BN_ROWS * CH;
        k_cvt_bf16<<<dim3(nx / (256 * 8)), dim3(256), 0, stream>>>(x, xb, nx);
        int nq = CH * QKV_C;
        k_cvt_bf16_t<<<dim3((nq + 255) / 256), dim3(256), 0, stream>>>(w_qkv, wqbT, CH, QKV_C);
        int np = CH * CH;
        k_cvt_bf16_t<<<dim3((np + 255) / 256), dim3(256), 0, stream>>>(w_proj, wpbT, CH, CH);
    }
    // 2) qkv projection: [65536x768]@[768x2304], async double-buffered staging
    k_gemm_qkv<<<dim3(QKV_C / 128, BN_ROWS / 256, 1), dim3(256), 0, stream>>>(xb, wqbT, qkvb);
    // 3) kv = gelu(scale * k^T v) per (b,h)
    k_kv<<<dim3(BATCH * NH), dim3(256), 0, stream>>>(qkvb, gkvb);
    // 4) attn = gelu(q) @ gkv per (b,h)
    k_gemm_attn<<<dim3(1, SEQ / 128, BATCH * NH), dim3(256), 0, stream>>>(qkvb, gkvb, attnb);
    // 5) out = attn @ w_proj + b_proj (fp32), async double-buffered staging
    k_gemm_proj<<<dim3(CH / 128, BN_ROWS / 256, 1), dim3(256), 0, stream>>>(attnb, wpbT, out, b_proj);
}
